// ImageEncoderViT_28381143892560
// MI455X (gfx1250) — matrix operand order repaired and checked by reading, not compiled, not run on hardware
//
#include <hip/hip_runtime.h>

typedef __bf16 bf16_t;
typedef __attribute__((ext_vector_type(16))) __bf16 bf16x16;
typedef __attribute__((ext_vector_type(8))) float floatx8;

#define TILE_M 64
#define TILE_N 128
#define TILE_K 64
#define LDS_PAD 8

// ---------------------------------------------------------------------------
// Block reductions (blockDim.x == 256 -> 8 waves of 32)
// ---------------------------------------------------------------------------
__device__ inline float blockSum(float v, float* sm) {
#pragma unroll
  for (int o = 16; o > 0; o >>= 1) v += __shfl_xor(v, o, 32);
  int lane = threadIdx.x & 31, wid = threadIdx.x >> 5;
  __syncthreads();
  if (lane == 0) sm[wid] = v;
  __syncthreads();
  float r = 0.f;
#pragma unroll
  for (int i = 0; i < 8; ++i) r += sm[i];
  return r;
}

__device__ inline float blockMax(float v, float* sm) {
#pragma unroll
  for (int o = 16; o > 0; o >>= 1) v = fmaxf(v, __shfl_xor(v, o, 32));
  int lane = threadIdx.x & 31, wid = threadIdx.x >> 5;
  __syncthreads();
  if (lane == 0) sm[wid] = v;
  __syncthreads();
  float r = -3.0e38f;
#pragma unroll
  for (int i = 0; i < 8; ++i) r = fmaxf(r, sm[i]);
  return r;
}

// ---------------------------------------------------------------------------
// Generic batched bf16 WMMA GEMM:  C[b] = act( alpha * A[b] x B[b] + bias + residual )
// A: [M,K] bf16 row-major (lda).   B element (n,k) at Bm[n*bN + k*bK].
// Block tile 64x128, K-step 64; 8 waves, each wave 32x32 => 8 WMMAs / K-tile.
// Staging paths: (1) fully-interior: async global->LDS b64 copies (ASYNCcnt),
// (2) row-guarded 8-byte register loads, (3) scalar for K-edge/strided-B.
// ---------------------------------------------------------------------------
__global__ __launch_bounds__(256)
void gemm_wmma_kernel(const bf16_t* __restrict__ A, const bf16_t* __restrict__ Bm,
                      const float* __restrict__ bias, const float* __restrict__ residual,
                      float* __restrict__ Cf, bf16_t* __restrict__ Cb,
                      int M, int N, int K,
                      long long sAb, long long sBb, long long sCb,
                      int lda, int bN, int bK, int ldc,
                      float alpha, int act) {
  __shared__ bf16_t As[TILE_M][TILE_K + LDS_PAD];
  __shared__ bf16_t Bs[TILE_N][TILE_K + LDS_PAD];

  const int tid   = threadIdx.x;
  const int lane  = tid & 31;
  const int wid   = tid >> 5;
  const int waveM = wid >> 2;   // 0..1  -> 32 rows each
  const int waveN = wid & 3;    // 0..3  -> 32 cols each
  const int batch = blockIdx.z;
  const int mBlock = blockIdx.y * TILE_M;
  const int nBlock = blockIdx.x * TILE_N;

  A  += (long long)batch * sAb;
  Bm += (long long)batch * sBb;

  floatx8 acc[2][2] = {{{}, {}}, {{}, {}}};

  const int halfSel = lane >> 4;  // 0/1
  const int l15 = lane & 15;

  // staging coordinates
  const int aRow = tid >> 2;            // 0..63
  const int aCol = (tid & 3) * 16;      // 0,16,32,48
  const int bRow = tid >> 1;            // 0..127
  const int bCol = (tid & 1) * 32;      // 0,32
  const int gmA = mBlock + aRow;
  const int gnB = nBlock + bRow;
  const bool vecOK = (bK == 1);
  const bool interior = vecOK && (mBlock + TILE_M <= M) && (nBlock + TILE_N <= N);

  const unsigned ldsA = (unsigned)(uintptr_t)&As[aRow][aCol];
  const unsigned ldsB = (unsigned)(uintptr_t)&Bs[bRow][bCol];

  for (int k0 = 0; k0 < K; k0 += TILE_K) {
    const bool fullK = (k0 + TILE_K <= K);
    if (interior && fullK) {
      // ---- async DMA staging: per-lane 8B copies straight into LDS ----
      unsigned long long gA =
          (unsigned long long)(uintptr_t)(A + (long long)gmA * lda + k0 + aCol);
      asm volatile(
          "global_load_async_to_lds_b64 %0, %1, off\n\t"
          "global_load_async_to_lds_b64 %0, %1, off offset:8\n\t"
          "global_load_async_to_lds_b64 %0, %1, off offset:16\n\t"
          "global_load_async_to_lds_b64 %0, %1, off offset:24"
          :: "v"(ldsA), "v"(gA) : "memory");
      unsigned long long gB =
          (unsigned long long)(uintptr_t)(Bm + (long long)gnB * bN + k0 + bCol);
      asm volatile(
          "global_load_async_to_lds_b64 %0, %1, off\n\t"
          "global_load_async_to_lds_b64 %0, %1, off offset:8\n\t"
          "global_load_async_to_lds_b64 %0, %1, off offset:16\n\t"
          "global_load_async_to_lds_b64 %0, %1, off offset:24\n\t"
          "global_load_async_to_lds_b64 %0, %1, off offset:32\n\t"
          "global_load_async_to_lds_b64 %0, %1, off offset:40\n\t"
          "global_load_async_to_lds_b64 %0, %1, off offset:48\n\t"
          "global_load_async_to_lds_b64 %0, %1, off offset:56"
          :: "v"(ldsB), "v"(gB) : "memory");
      if (k0 + TILE_K < K) {
        __builtin_prefetch(A + (long long)gmA * lda + k0 + TILE_K + aCol, 0, 1);
        __builtin_prefetch(Bm + (long long)gnB * bN + k0 + TILE_K + bCol, 0, 1);
      }
      asm volatile("s_wait_asynccnt 0x0" ::: "memory");
    } else if (vecOK && fullK) {
      // ---- vector staging: 8-byte loads, only per-row guards ----
      {
        const bf16_t* src = A + (long long)gmA * lda + k0 + aCol;
#pragma unroll
        for (int j = 0; j < 4; ++j) {
          unsigned long long v = 0ull;
          if (gmA < M) v = *(const unsigned long long*)(src + j * 4);
          *(unsigned long long*)&As[aRow][aCol + j * 4] = v;
        }
      }
      {
        const bf16_t* src = Bm + (long long)gnB * bN + k0 + bCol;
#pragma unroll
        for (int j = 0; j < 8; ++j) {
          unsigned long long v = 0ull;
          if (gnB < N) v = *(const unsigned long long*)(src + j * 4);
          *(unsigned long long*)&Bs[bRow][bCol + j * 4] = v;
        }
      }
    } else {
      // ---- scalar guarded staging (K edge tiles / strided B only) ----
#pragma unroll 4
      for (int j = 0; j < 16; ++j) {
        int gk = k0 + aCol + j;
        bf16_t v = (bf16_t)0.f;
        if (gmA < M && gk < K) v = A[(long long)gmA * lda + gk];
        As[aRow][aCol + j] = v;
      }
#pragma unroll 4
      for (int j = 0; j < 32; ++j) {
        int gk = k0 + bCol + j;
        bf16_t v = (bf16_t)0.f;
        if (gnB < N && gk < K) v = Bm[(long long)gnB * bN + (long long)gk * bK];
        Bs[bRow][bCol + j] = v;
      }
    }
    __syncthreads();

    // ---- compute: 2 k-subtiles x (2 A-frags x 2 B-frags) WMMAs ----
    const int rA0 = waveM * 32 + l15;
    const int rA1 = rA0 + 16;
    const int c0  = waveN * 32 + l15;
#pragma unroll
    for (int kk = 0; kk < TILE_K; kk += 32) {
      bf16x16 aF0, aF1, bF0, bF1;
      {
        int kb = kk + halfSel * 8;   // lanes 0-15: K 0-7/16-23, lanes 16-31: 8-15/24-31
#pragma unroll
        for (int vv = 0; vv < 8; ++vv) {
          int k = kb + (vv & 3) * 2 + (vv >> 2) * 16;
          aF0[2 * vv]     = As[rA0][k];
          aF0[2 * vv + 1] = As[rA0][k + 1];
          aF1[2 * vv]     = As[rA1][k];
          aF1[2 * vv + 1] = As[rA1][k + 1];
        }
        int kbb = kk + halfSel * 8;  // lanes 0-15: K 0-7/16-23, lanes 16-31: 8-15/24-31
#pragma unroll
        for (int vv = 0; vv < 8; ++vv) {
          int k = kbb + (vv & 3) * 2 + (vv >> 2) * 16;
          bF0[2 * vv]     = Bs[c0][k];
          bF0[2 * vv + 1] = Bs[c0][k + 1];
          bF1[2 * vv]     = Bs[c0 + 16][k];
          bF1[2 * vv + 1] = Bs[c0 + 16][k + 1];
        }
      }
      acc[0][0] = __builtin_amdgcn_wmma_f32_16x16x32_bf16(false, aF0, false, bF0, (short)0, acc[0][0], false, false);
      acc[0][1] = __builtin_amdgcn_wmma_f32_16x16x32_bf16(false, aF0, false, bF1, (short)0, acc[0][1], false, false);
      acc[1][0] = __builtin_amdgcn_wmma_f32_16x16x32_bf16(false, aF1, false, bF0, (short)0, acc[1][0], false, false);
      acc[1][1] = __builtin_amdgcn_wmma_f32_16x16x32_bf16(false, aF1, false, bF1, (short)0, acc[1][1], false, false);
    }
    __syncthreads();
  }

  // ---- epilogue (C layout: lane 0-15 -> M 0-7, lane 16-31 -> M 8-15) ----
  long long cBase = (long long)batch * sCb;
#pragma unroll
  for (int smt = 0; smt < 2; ++smt) {
#pragma unroll
    for (int vv = 0; vv < 8; ++vv) {
      int m = mBlock + waveM * 32 + smt * 16 + halfSel * 8 + vv;
      if (m >= M) continue;
#pragma unroll
      for (int t = 0; t < 2; ++t) {
        int n = nBlock + waveN * 32 + t * 16 + l15;
        if (n >= N) continue;
        float val = acc[smt][t][vv] * alpha;
        if (bias) val += bias[n];
        long long off = cBase + (long long)m * ldc + n;
        if (residual) val += residual[off];
        if (act == 1) val = 0.5f * val * (1.0f + erff(val * 0.70710678118654752f));
        if (Cf) Cf[off] = val;
        if (Cb) Cb[off] = (bf16_t)val;
      }
    }
  }
}

// ---------------------------------------------------------------------------
// Elementwise / data-movement kernels
// ---------------------------------------------------------------------------
__global__ void f32_to_bf16_kernel(const float* __restrict__ s, bf16_t* __restrict__ d, long long n) {
  long long i = (long long)blockIdx.x * 256 + threadIdx.x;
  if (i < n) d[i] = (bf16_t)s[i];
}

__global__ void fill_bf16_kernel(bf16_t* __restrict__ p, long long n) {
  long long i = (long long)blockIdx.x * 256 + threadIdx.x;
  if (i < n) p[i] = (bf16_t)0.f;
}

__global__ void add_pos_kernel(float* __restrict__ x, const float* __restrict__ pos, long long n) {
  long long i = (long long)blockIdx.x * 256 + threadIdx.x;
  if (i < n) x[i] += pos[i % (1024LL * 768LL)];
}

// x [2,3,512,512] f32 -> im2col bf16 [2048, 768] (k = c*256 + py*16 + px)
__global__ void im2col_patch_kernel(const float* __restrict__ x, bf16_t* __restrict__ out) {
  int row = blockIdx.x;                // b*1024 + gy*32 + gx
  int bI = row >> 10;
  int g = row & 1023;
  int gy = g >> 5, gx = g & 31;
  for (int k = threadIdx.x; k < 768; k += blockDim.x) {
    int c = k >> 8;
    int rem = k & 255;
    int py = rem >> 4, px = rem & 15;
    long long src = (((long long)(bI * 3 + c) * 512) + gy * 16 + py) * 512 + gx * 16 + px;
    out[(long long)row * 768 + k] = (bf16_t)x[src];
  }
}

// LayerNorm over C, f32 in -> bf16 out; scatter=1 writes into padded window layout
__global__ __launch_bounds__(256)
void layernorm_kernel(const float* __restrict__ x, const float* __restrict__ w,
                      const float* __restrict__ b, bf16_t* __restrict__ out,
                      int C2, float eps, int scatter) {
  __shared__ float sm[8];
  int row = blockIdx.x;
  const float* xr = x + (long long)row * C2;
  float s = 0.f, s2 = 0.f;
  for (int c = threadIdx.x; c < C2; c += blockDim.x) { float v = xr[c]; s += v; s2 += v * v; }
  s  = blockSum(s, sm);
  s2 = blockSum(s2, sm);
  float mean = s / C2;
  float var  = s2 / C2 - mean * mean;
  float inv  = rsqrtf(var + eps);
  long long dstRow = row;
  if (scatter) {                        // (b,h,w) -> window batch (b*9 + wy*3 + wx), token py*14+px
    int bI = row >> 10; int hw = row & 1023;
    int h = hw >> 5, wq = hw & 31;
    int wb = bI * 9 + (h / 14) * 3 + (wq / 14);
    int t  = (h % 14) * 14 + (wq % 14);
    dstRow = (long long)wb * 196 + t;
  }
  for (int c = threadIdx.x; c < C2; c += blockDim.x)
    out[dstRow * C2 + c] = (bf16_t)((xr[c] - mean) * inv * w[c] + b[c]);
}

// qkv f32 [nB*T, 2304] -> q/k bf16 [(nB*12), T, 64], v transposed -> [(nB*12), 64, T]
__global__ void split_qkv_kernel(const float* __restrict__ qkv, bf16_t* __restrict__ q,
                                 bf16_t* __restrict__ k, bf16_t* __restrict__ vT,
                                 int nB, int T) {
  long long i = (long long)blockIdx.x * 256 + threadIdx.x;
  long long tot = (long long)nB * T * 768;
  if (i >= tot) return;
  int dd = (int)(i & 63);
  int hd = (int)((i >> 6) % 12);
  long long rem = i / 768;
  int t = (int)(rem % T);
  int bI = (int)(rem / T);
  long long srcRow = (long long)bI * T + t;
  int col = hd * 64 + dd;
  long long bhBase = ((long long)bI * 12 + hd);
  long long dst = (bhBase * T + t) * 64 + dd;
  q[dst] = (bf16_t)qkv[srcRow * 2304 + col];
  k[dst] = (bf16_t)qkv[srcRow * 2304 + 768 + col];
  vT[(bhBase * 64 + dd) * T + t] = (bf16_t)qkv[srcRow * 2304 + 1536 + col];
}

// Softmax with decomposed rel-pos bias; one block per (query row, batch*head)
__global__ __launch_bounds__(256)
void softmax_relpos_kernel(float* __restrict__ scores, bf16_t* __restrict__ attn,
                           const bf16_t* __restrict__ q, const float* __restrict__ relH,
                           const float* __restrict__ relW, int T, int Hs, int Ws) {
  __shared__ float qs[64];
  __shared__ float dh[32];
  __shared__ float dw[32];
  __shared__ float sm[8];
  int row = blockIdx.x;
  int bh  = blockIdx.y;
  long long base = ((long long)bh * T + row) * T;
  const bf16_t* qrow = q + ((long long)bh * T + row) * 64;
  int tid = threadIdx.x;
  if (tid < 64) qs[tid] = (float)qrow[tid];
  __syncthreads();
  int qh = row / Ws, qw = row - qh * Ws;
  if (tid < Hs) {
    const float* rp = relH + (long long)(qh - tid + Hs - 1) * 64;
    float s = 0.f;
#pragma unroll 8
    for (int c = 0; c < 64; ++c) s += qs[c] * rp[c];
    dh[tid] = s;
  }
  if (tid >= 64 && tid < 64 + Ws) {
    int kw = tid - 64;
    const float* rp = relW + (long long)(qw - kw + Ws - 1) * 64;
    float s = 0.f;
#pragma unroll 8
    for (int c = 0; c < 64; ++c) s += qs[c] * rp[c];
    dw[kw] = s;
  }
  __syncthreads();
  float mx = -3.0e38f;
  for (int j = tid; j < T; j += blockDim.x) {
    int kh = j / Ws, kw = j - kh * Ws;
    float s = scores[base + j] + dh[kh] + dw[kw];
    scores[base + j] = s;
    mx = fmaxf(mx, s);
  }
  mx = blockMax(mx, sm);
  float sum = 0.f;
  for (int j = tid; j < T; j += blockDim.x) {
    float e = __expf(scores[base + j] - mx);
    scores[base + j] = e;
    sum += e;
  }
  sum = blockSum(sum, sm);
  float rinv = 1.0f / sum;
  for (int j = tid; j < T; j += blockDim.x)
    attn[base + j] = (bf16_t)(scores[base + j] * rinv);
}

// Merge heads (+ reverse window partition) -> bf16 [2048, 768]
__global__ void merge_heads_kernel(const float* __restrict__ ao, bf16_t* __restrict__ out,
                                   int windowed) {
  int row = blockIdx.x;                // b*1024 + h*32 + w
  int bI = row >> 10;
  int hw = row & 1023;
  int h = hw >> 5, w = hw & 31;
  for (int c = threadIdx.x; c < 768; c += blockDim.x) {
    int hd = c >> 6, dd = c & 63;
    long long src;
    if (windowed) {
      int wb = bI * 9 + (h / 14) * 3 + (w / 14);
      int t  = (h % 14) * 14 + (w % 14);
      src = (((long long)wb * 12 + hd) * 196 + t) * 64 + dd;
    } else {
      src = (((long long)bI * 12 + hd) * 1024 + hw) * 64 + dd;
    }
    out[(long long)row * 768 + c] = (bf16_t)ao[src];
  }
}

// LayerNorm2d over 256 channels per pixel; optional bf16 [row,c] and/or NCHW f32 out
__global__ __launch_bounds__(256)
void ln2d_kernel(const float* __restrict__ x, const float* __restrict__ w,
                 const float* __restrict__ b, bf16_t* __restrict__ outBf,
                 float* __restrict__ outT, int C2, float eps) {
  __shared__ float sm[8];
  int row = blockIdx.x;
  int tid = threadIdx.x;
  float v = x[(long long)row * C2 + tid];
  float s  = blockSum(v, sm);
  float s2 = blockSum(v * v, sm);
  float mean = s / C2;
  float var  = s2 / C2 - mean * mean;
  float inv  = rsqrtf(var + eps);
  float r = (v - mean) * inv * w[tid] + b[tid];
  if (outBf) outBf[(long long)row * C2 + tid] = (bf16_t)r;
  if (outT) {
    int bI = row >> 10;
    int hw = row & 1023;
    outT[((long long)bI * C2 + tid) * 1024 + hw] = r;
  }
}

// bf16 [b,32,32,256] -> im2col bf16 [2048, 2304] for 3x3 SAME conv
__global__ void im2col3_kernel(const bf16_t* __restrict__ in, bf16_t* __restrict__ out) {
  int row = blockIdx.x;
  int bI = row >> 10;
  int hw = row & 1023;
  int h = hw >> 5, w = hw & 31;
  for (int k = threadIdx.x; k < 2304; k += blockDim.x) {
    int i  = k / 9;
    int r9 = k % 9;
    int ky = r9 / 3, kx = r9 % 3;
    int hh = h + ky - 1, ww = w + kx - 1;
    bf16_t v = (bf16_t)0.f;
    if (hh >= 0 && hh < 32 && ww >= 0 && ww < 32)
      v = in[(((long long)bI * 32 + hh) * 32 + ww) * 256 + i];
    out[(long long)row * 2304 + k] = v;
  }
}

// ---------------------------------------------------------------------------
// Host orchestration
// ---------------------------------------------------------------------------
extern "C" void kernel_launch(void* const* d_in, const int* in_sizes, int n_in,
                              void* d_out, int out_size, void* d_ws, size_t ws_size,
                              hipStream_t stream) {
  (void)in_sizes; (void)n_in; (void)out_size; (void)ws_size;

  const int Cd = 768;
  const long long rowsImg = 2048;          // 2 * 32 * 32
  const int TW = 196, WB = 18, BHw = 216;  // windowed attention batches
  const int Tg = 1024, BHg = 24;           // global attention batches
  const long long MW = (long long)WB * TW; // 3528 padded window tokens

  // ---- input pointers (setup_inputs dict order, block dicts flattened) ----
  const float* xIn     = (const float*)d_in[0];
  const float* patchW  = (const float*)d_in[1];
  const float* patchB  = (const float*)d_in[2];
  const float* posEmb  = (const float*)d_in[3];
  const int WINB = 4, GLOB = 18;           // param bases; order within dict:
  // 0 n1_w, 1 n1_b, 2 qkv_w, 3 qkv_b, 4 proj_w, 5 proj_b, 6 rel_h, 7 rel_w,
  // 8 n2_w, 9 n2_b, 10 w1, 11 b1, 12 w2, 13 b2
  const float* neckW1  = (const float*)d_in[32];
  const float* ln1w    = (const float*)d_in[33];
  const float* ln1b    = (const float*)d_in[34];
  const float* neckW2  = (const float*)d_in[35];
  const float* ln2w    = (const float*)d_in[36];
  const float* ln2b    = (const float*)d_in[37];

  // ---- workspace carve ----
  char* ws = (char*)d_ws;
  size_t off = 0;
  auto alloc = [&](size_t bytes) -> void* {
    void* p = ws + off;
    off = (off + bytes + 255) & ~(size_t)255;
    return p;
  };
  const long long nQKVW = 2LL * 2304 * 768, nPROJ = 2LL * 768 * 768;
  const long long nW1 = 2LL * 3072 * 768, nW2 = 2LL * 768 * 3072;
  bf16_t* qkvWbf[2]; bf16_t* projWbf[2]; bf16_t* w1bf[2]; bf16_t* w2bf[2];
  for (int g = 0; g < 2; ++g) {
    qkvWbf[g] = (bf16_t*)alloc(nQKVW * 2);
    projWbf[g] = (bf16_t*)alloc(nPROJ * 2);
    w1bf[g] = (bf16_t*)alloc(nW1 * 2);
    w2bf[g] = (bf16_t*)alloc(nW2 * 2);
  }
  bf16_t* patchWbf = (bf16_t*)alloc(768LL * 768 * 2);
  bf16_t* neckW1bf = (bf16_t*)alloc(256LL * 768 * 2);
  bf16_t* neckW2bf = (bf16_t*)alloc(256LL * 2304 * 2);

  float*  xStream = (float*)alloc(rowsImg * Cd * 4);
  bf16_t* lnBuf   = (bf16_t*)alloc(MW * Cd * 2);                 // windowed LN / plain LN / neck A
  float*  qkvF    = (float*)alloc(MW * 3 * Cd * 4);
  const long long nQKVmax = (long long)BHw * TW * 64;            // >= global size
  bf16_t* qB  = (bf16_t*)alloc(nQKVmax * 2);
  bf16_t* kB  = (bf16_t*)alloc(nQKVmax * 2);
  bf16_t* vTB = (bf16_t*)alloc(nQKVmax * 2);
  const long long nScore = (long long)BHg * Tg * Tg;             // 24M > 216*196*196
  float*  scoresF = (float*)alloc(nScore * 4);
  bf16_t* attnBf  = (bf16_t*)alloc(nScore * 2);
  float*  attnOutF = (float*)alloc(nQKVmax * 4);
  bf16_t* mergedBf = (bf16_t*)alloc(rowsImg * Cd * 2);
  bf16_t* mlpMid  = (bf16_t*)alloc(rowsImg * 4 * Cd * 2);
  float*  neckF   = (float*)alloc(rowsImg * 256 * 4);
  bf16_t* neckBf  = (bf16_t*)alloc(rowsImg * 256 * 2);
  bf16_t* colBf   = (bf16_t*)alloc(rowsImg * 2304 * 2);

  auto cvt = [&](const float* s, bf16_t* d, long long n) {
    f32_to_bf16_kernel<<<(unsigned)((n + 255) / 256), 256, 0, stream>>>(s, d, n);
  };
  auto gemm = [&](const bf16_t* A, const bf16_t* Bw, const float* bias, const float* resid,
                  float* Cf, bf16_t* Cb, int M, int N, int K,
                  long long sA, long long sB, long long sC,
                  int lda, int bNs, int bKs, int ldc, int batch, float alpha, int act) {
    dim3 g((N + TILE_N - 1) / TILE_N, (M + TILE_M - 1) / TILE_M, batch);
    gemm_wmma_kernel<<<g, 256, 0, stream>>>(A, Bw, bias, resid, Cf, Cb, M, N, K,
                                            sA, sB, sC, lda, bNs, bKs, ldc, alpha, act);
  };

  // ---- weight conversion to bf16 ----
  for (int g = 0; g < 2; ++g) {
    int base = g ? GLOB : WINB;
    cvt((const float*)d_in[base + 2], qkvWbf[g], nQKVW);
    cvt((const float*)d_in[base + 4], projWbf[g], nPROJ);
    cvt((const float*)d_in[base + 10], w1bf[g], nW1);
    cvt((const float*)d_in[base + 12], w2bf[g], nW2);
  }
  cvt(patchW, patchWbf, 768LL * 768);
  cvt(neckW1, neckW1bf, 256LL * 768);
  cvt(neckW2, neckW2bf, 256LL * 2304);

  // ---- patch embedding: im2col -> GEMM(+bias) -> +pos_embed ----
  im2col_patch_kernel<<<(unsigned)rowsImg, 256, 0, stream>>>(xIn, colBf);
  gemm(colBf, patchWbf, patchB, nullptr, xStream, nullptr,
       (int)rowsImg, Cd, Cd, 0, 0, 0, Cd, Cd, 1, Cd, 1, 1.f, 0);
  add_pos_kernel<<<(unsigned)((rowsImg * Cd + 255) / 256), 256, 0, stream>>>(
      xStream, posEmb, rowsImg * Cd);

  // ---- transformer blocks: depth 4, GLOBAL = (1,3) ----
  int wi = 0, gi = 0;
  for (int d = 0; d < 4; ++d) {
    bool glob = (d == 1 || d == 3);
    int gsel = glob ? 1 : 0;
    int i = glob ? gi++ : wi++;
    int base = glob ? GLOB : WINB;
    const float* n1w  = (const float*)d_in[base + 0] + (long long)i * Cd;
    const float* n1b  = (const float*)d_in[base + 1] + (long long)i * Cd;
    const float* qkvb = (const float*)d_in[base + 3] + (long long)i * 3 * Cd;
    const float* pjb  = (const float*)d_in[base + 5] + (long long)i * Cd;
    int relL = glob ? 63 : 27;
    const float* relH = (const float*)d_in[base + 6] + (long long)i * relL * 64;
    const float* relW = (const float*)d_in[base + 7] + (long long)i * relL * 64;
    const float* n2w  = (const float*)d_in[base + 8] + (long long)i * Cd;
    const float* n2b  = (const float*)d_in[base + 9] + (long long)i * Cd;
    const float* b1   = (const float*)d_in[base + 11] + (long long)i * 4 * Cd;
    const float* b2   = (const float*)d_in[base + 13] + (long long)i * Cd;
    const bf16_t* qkvW = qkvWbf[gsel] + (long long)i * 3 * Cd * Cd;
    const bf16_t* pjW  = projWbf[gsel] + (long long)i * Cd * Cd;
    const bf16_t* m1W  = w1bf[gsel] + (long long)i * 4 * Cd * Cd;
    const bf16_t* m2W  = w2bf[gsel] + (long long)i * Cd * 4 * Cd;

    const int T  = glob ? Tg : TW;
    const int BH = glob ? BHg : BHw;
    const int nB = glob ? 2 : WB;
    const int Hs = glob ? 32 : 14;
    const long long Mq = glob ? rowsImg : MW;

    if (!glob) {
      fill_bf16_kernel<<<(unsigned)((MW * Cd + 255) / 256), 256, 0, stream>>>(lnBuf, MW * Cd);
      layernorm_kernel<<<(unsigned)rowsImg, 256, 0, stream>>>(xStream, n1w, n1b, lnBuf, Cd, 1e-5f, 1);
    } else {
      layernorm_kernel<<<(unsigned)rowsImg, 256, 0, stream>>>(xStream, n1w, n1b, lnBuf, Cd, 1e-5f, 0);
    }
    gemm(lnBuf, qkvW, qkvb, nullptr, qkvF, nullptr,
         (int)Mq, 3 * Cd, Cd, 0, 0, 0, Cd, Cd, 1, 3 * Cd, 1, 1.f, 0);
    long long nSplit = (long long)nB * T * Cd;
    split_qkv_kernel<<<(unsigned)((nSplit + 255) / 256), 256, 0, stream>>>(qkvF, qB, kB, vTB, nB, T);
    gemm(qB, kB, nullptr, nullptr, scoresF, nullptr, T, T, 64,
         (long long)T * 64, (long long)T * 64, (long long)T * T,
         64, 64, 1, T, BH, 0.125f, 0);
    softmax_relpos_kernel<<<dim3(T, BH), 256, 0, stream>>>(scoresF, attnBf, qB, relH, relW, T, Hs, Hs);
    gemm(attnBf, vTB, nullptr, nullptr, attnOutF, nullptr, T, 64, T,
         (long long)T * T, (long long)T * 64, (long long)T * 64,
         T, T, 1, 64, BH, 1.f, 0);
    merge_heads_kernel<<<(unsigned)rowsImg, 256, 0, stream>>>(attnOutF, mergedBf, glob ? 0 : 1);

    // proj + residual (x = shortcut + proj(attn))
    gemm(mergedBf, pjW, pjb, xStream, xStream, nullptr,
         (int)rowsImg, Cd, Cd, 0, 0, 0, Cd, Cd, 1, Cd, 1, 1.f, 0);
    // MLP: ln2 -> gelu(GEMM1) -> GEMM2 + residual
    layernorm_kernel<<<(unsigned)rowsImg, 256, 0, stream>>>(xStream, n2w, n2b, lnBuf, Cd, 1e-5f, 0);
    gemm(lnBuf, m1W, b1, nullptr, nullptr, mlpMid,
         (int)rowsImg, 4 * Cd, Cd, 0, 0, 0, Cd, Cd, 1, 4 * Cd, 1, 1.f, 1);
    gemm(mlpMid, m2W, b2, xStream, xStream, nullptr,
         (int)rowsImg, Cd, 4 * Cd, 0, 0, 0, 4 * Cd, 4 * Cd, 1, Cd, 1, 1.f, 0);
  }

  // ---- neck ----
  cvt(xStream, lnBuf, rowsImg * Cd);
  gemm(lnBuf, neckW1bf, nullptr, nullptr, neckF, nullptr,
       (int)rowsImg, 256, Cd, 0, 0, 0, Cd, Cd, 1, 256, 1, 1.f, 0);
  ln2d_kernel<<<(unsigned)rowsImg, 256, 0, stream>>>(neckF, ln1w, ln1b, neckBf, nullptr, 256, 1e-6f);
  im2col3_kernel<<<(unsigned)rowsImg, 256, 0, stream>>>(neckBf, colBf);
  gemm(colBf, neckW2bf, nullptr, nullptr, neckF, nullptr,
       (int)rowsImg, 256, 2304, 0, 0, 0, 2304, 2304, 1, 256, 1, 1.f, 0);
  ln2d_kernel<<<(unsigned)rowsImg, 256, 0, stream>>>(neckF, ln2w, ln2b, nullptr, (float*)d_out, 256, 1e-6f);
}
